// SelfAttention_65481071409934
// MI455X (gfx1250) — matrix operand order repaired and checked by reading, not compiled, not run on hardware
//
#include <hip/hip_runtime.h>

typedef _Float16 v16h __attribute__((ext_vector_type(16)));
typedef _Float16 v8h  __attribute__((ext_vector_type(8)));
typedef float    v8f  __attribute__((ext_vector_type(8)));

#define B_ 4
#define S_ 4096
#define D_ 64
#define BC 64               // kv rows per block (4 M-tiles of 16)

union FragU { v16h v; v8h h[2]; };

// ---- WMMA: D = A(16x32 f16) * B(32x16 f16) + C(16x16 f32) -----------------
__device__ __forceinline__ v8f wmma_f16(v16h a, v16h b, v8f c) {
    return __builtin_amdgcn_wmma_f32_16x16x32_f16(
        /*neg_a=*/false, a, /*neg_b=*/false, b,
        /*c_mod=*/(short)0, c, /*reuse_a=*/false, /*reuse_b=*/false);
}

// A fragment (16x32, MxK) from row-major X[M][ld]:
// lane m = lane&15; halves = X[m][c0 + (lane>>4)*8 + i], X[m][c0+16+(lane>>4)*8 + i]
__device__ __forceinline__ v16h load_fragA(const _Float16* __restrict__ base,
                                           int ld, int lane, int c0) {
    const _Float16* p = base + (size_t)(lane & 15) * ld + c0 + ((lane >> 4) << 3);
    FragU f;
    f.h[0] = *(const v8h*)(p);
    f.h[1] = *(const v8h*)(p + 16);
    return f.v;
}

// B fragment (32x16, KxN) where B = X^T for row-major X[N][ld]:
// lane n = lane&15; halves = X[n][c0 + (lane>>4)*8 + i], X[n][c0+16+(lane>>4)*8 + i]
__device__ __forceinline__ v16h load_fragBT(const _Float16* __restrict__ base,
                                            int ld, int lane, int c0) {
    const _Float16* p = base + (size_t)(lane & 15) * ld + c0 + ((lane >> 4) << 3);
    FragU f;
    f.h[0] = *(const v8h*)(p);
    f.h[1] = *(const v8h*)(p + 16);
    return f.v;
}

// ---------------------------------------------------------------------------
// Kernel 1: Q = (X*rot)/sqrt(D), K = X*ent (f16), Vt = X^T (f16, [B][D][S])
// ---------------------------------------------------------------------------
__global__ __launch_bounds__(256) void
attn_prep_kernel(const float* __restrict__ inp, const float* __restrict__ rot,
                 const float* __restrict__ ent, _Float16* __restrict__ Qh,
                 _Float16* __restrict__ Kh, _Float16* __restrict__ Vt) {
    int idx = blockIdx.x * 256 + threadIdx.x;    // over B*S*D
    int e   = idx & (D_ - 1);
    int row = idx >> 6;                           // b*S + s
    const float* x = inp + (size_t)row * D_;
    float q = 0.f, k = 0.f;
#pragma unroll 8
    for (int d = 0; d < D_; ++d) {
        float xv = x[d];
        q = fmaf(xv, rot[d * D_ + e], q);
        k = fmaf(xv, ent[d * D_ + e], k);
    }
    Qh[idx] = (_Float16)(q * 0.125f);            // fold 1/sqrt(64) into Q
    Kh[idx] = (_Float16)k;
    int b = row >> 12;
    int s = row & (S_ - 1);
    Vt[((size_t)b * D_ + e) * S_ + s] = (_Float16)x[e];
}

// ---------------------------------------------------------------------------
// Kernel 2: flash attention, transposed-score formulation.
// One wave = one 16-row Q tile; softmax reductions are in-lane + one xor-16.
// S^T C-fragments ARE the P A-fragments (no LDS relayout needed).
// ---------------------------------------------------------------------------
__global__ __launch_bounds__(128) void
flash_attn_kernel(const _Float16* __restrict__ Qh, const _Float16* __restrict__ Kh,
                  const _Float16* __restrict__ Vt, float* __restrict__ out) {
    const int lane = threadIdx.x & 31;
    const int wave = threadIdx.x >> 5;
    const int tile = blockIdx.x * 4 + wave;       // 0 .. B*S/16-1
    const int b    = tile >> 8;                   // 256 tiles per batch
    const int q0   = (tile & 255) << 4;

    const _Float16* Qb = Qh + (size_t)b * S_ * D_ + (size_t)q0 * D_;
    const _Float16* Kb = Kh + (size_t)b * S_ * D_;
    const _Float16* Vb = Vt + (size_t)b * D_ * S_;
    float*          Ob = out + ((size_t)b * S_ + q0) * D_;

    // Q^T as B-operand (resident): B[k][n] = Q[n][k]
    const v16h qbt0 = load_fragBT(Qb, D_, lane, 0);
    const v16h qbt1 = load_fragBT(Qb, D_, lane, 32);

    v8f o0 = {}, o1 = {}, o2 = {}, o3 = {};       // O: 16 x 64 (4 d-tiles)
    float m = -3.0e38f, l = 0.f;                  // per-lane (q row = lane&15)
    const int rbase = (lane >> 4) << 3;

    for (int kv = 0; kv < S_; kv += BC) {
        const _Float16* Kblk = Kb + (size_t)kv * D_;
        // ---- S^T (64x16) = K_blk (64x64) x Q^T (64x16), 4 M-tiles ----
        v8f st0 = {}, st1 = {}, st2 = {}, st3 = {};
        st0 = wmma_f16(load_fragA(Kblk,           D_, lane, 0),  qbt0, st0);
        st0 = wmma_f16(load_fragA(Kblk,           D_, lane, 32), qbt1, st0);
        st1 = wmma_f16(load_fragA(Kblk + 16 * D_, D_, lane, 0),  qbt0, st1);
        st1 = wmma_f16(load_fragA(Kblk + 16 * D_, D_, lane, 32), qbt1, st1);
        st2 = wmma_f16(load_fragA(Kblk + 32 * D_, D_, lane, 0),  qbt0, st2);
        st2 = wmma_f16(load_fragA(Kblk + 32 * D_, D_, lane, 32), qbt1, st2);
        st3 = wmma_f16(load_fragA(Kblk + 48 * D_, D_, lane, 0),  qbt0, st3);
        st3 = wmma_f16(load_fragA(Kblk + 48 * D_, D_, lane, 32), qbt1, st3);

        if (kv + BC < S_) {   // prefetch next KV block (K: 8KB, Vt: 64 lines)
            const _Float16* kn = Kb + (size_t)(kv + BC) * D_;
            __builtin_prefetch(kn + lane * 64, 0, 0);
            __builtin_prefetch(kn + 2048 + lane * 64, 0, 0);
            __builtin_prefetch(Vb + (size_t)(2 * lane) * S_ + kv + BC, 0, 0);
            __builtin_prefetch(Vb + (size_t)(2 * lane + 1) * S_ + kv + BC, 0, 0);
        }

        // ---- online softmax: in-lane over 32 values + one xor-16 merge ----
        float mx = st0[0];
#pragma unroll
        for (int r = 0; r < 8; ++r) {
            mx = fmaxf(mx, fmaxf(fmaxf(st0[r], st1[r]), fmaxf(st2[r], st3[r])));
        }
        mx = fmaxf(mx, __shfl_xor(mx, 16, 32));
        float mn    = fmaxf(m, mx);
        float alpha = __expf(m - mn);
        m = mn;

        v16h pa0, pa1;                 // P A-fragments (K=0..31, 32..63)
        float rs = 0.f;
#pragma unroll
        for (int r = 0; r < 8; ++r) {
            float e0 = __expf(st0[r] - mn);
            float e1 = __expf(st1[r] - mn);
            float e2 = __expf(st2[r] - mn);
            float e3 = __expf(st3[r] - mn);
            rs += (e0 + e1) + (e2 + e3);
            pa0[r]     = (_Float16)e0;
            pa0[8 + r] = (_Float16)e1;
            pa1[r]     = (_Float16)e2;
            pa1[8 + r] = (_Float16)e3;
        }
        rs += __shfl_xor(rs, 16, 32);
        l = l * alpha + rs;

        // ---- rescale O by per-row alpha (row M = rbase + r) ----
#pragma unroll
        for (int r = 0; r < 8; ++r) {
            float ar = __shfl(alpha, rbase + r, 32);
            o0[r] *= ar; o1[r] *= ar; o2[r] *= ar; o3[r] *= ar;
        }

        // ---- O(16x64) += P(16x64) x V(64x64), B-operand from Vt rows ----
        o0 = wmma_f16(pa0, load_fragBT(Vb,           S_, lane, kv),      o0);
        o0 = wmma_f16(pa1, load_fragBT(Vb,           S_, lane, kv + 32), o0);
        o1 = wmma_f16(pa0, load_fragBT(Vb + 16 * S_, S_, lane, kv),      o1);
        o1 = wmma_f16(pa1, load_fragBT(Vb + 16 * S_, S_, lane, kv + 32), o1);
        o2 = wmma_f16(pa0, load_fragBT(Vb + 32 * S_, S_, lane, kv),      o2);
        o2 = wmma_f16(pa1, load_fragBT(Vb + 32 * S_, S_, lane, kv + 32), o2);
        o3 = wmma_f16(pa0, load_fragBT(Vb + 48 * S_, S_, lane, kv),      o3);
        o3 = wmma_f16(pa1, load_fragBT(Vb + 48 * S_, S_, lane, kv + 32), o3);
    }

    // ---- epilogue: O / l (per-row), f32 stores per the C/D VGPR map ----
    const int n = lane & 15;
#pragma unroll
    for (int r = 0; r < 8; ++r) {
        float lr  = __shfl(l, rbase + r, 32);
        float inv = 1.0f / lr;
        float* orow = Ob + (size_t)(rbase + r) * D_ + n;
        orow[0]  = o0[r] * inv;
        orow[16] = o1[r] * inv;
        orow[32] = o2[r] * inv;
        orow[48] = o3[r] * inv;
    }
}

// ---------------------------------------------------------------------------
extern "C" void kernel_launch(void* const* d_in, const int* in_sizes, int n_in,
                              void* d_out, int out_size, void* d_ws, size_t ws_size,
                              hipStream_t stream) {
    const float* inp = (const float*)d_in[0];
    const float* rot = (const float*)d_in[1];
    const float* ent = (const float*)d_in[2];
    float* out = (float*)d_out;

    _Float16* Qh = (_Float16*)d_ws;
    _Float16* Kh = Qh + (size_t)B_ * S_ * D_;
    _Float16* Vt = Kh + (size_t)B_ * S_ * D_;

    attn_prep_kernel<<<(B_ * S_ * D_) / 256, 256, 0, stream>>>(inp, rot, ent, Qh, Kh, Vt);
    flash_attn_kernel<<<(B_ * S_ / 16) / 4, 128, 0, stream>>>(Qh, Kh, Vt, out);
}
